// GraphTransformer_Embs_58205396795519
// MI455X (gfx1250) — matrix operand order repaired and checked by reading, not compiled, not run on hardware
//
#include <hip/hip_runtime.h>

typedef __attribute__((ext_vector_type(16))) _Float16 v16h;
typedef __attribute__((ext_vector_type(8)))  _Float16 v8h;
typedef __attribute__((ext_vector_type(8)))  float    v8f;
typedef __attribute__((ext_vector_type(4)))  float    v4f;

#define HDIM 64
#define NHEAD 4

// order-preserving float <-> uint encoding for atomicMax-based segment max
__device__ __forceinline__ unsigned encOrd(float f) {
    unsigned u = __float_as_uint(f);
    return (u & 0x80000000u) ? ~u : (u | 0x80000000u);
}
__device__ __forceinline__ float decOrd(unsigned u) {
    return __uint_as_float((u & 0x80000000u) ? (u & 0x7FFFFFFFu) : ~u);
}
#define NEG_INF_ENC 0x007FFFFFu   // encOrd(-inf)

// ---------------- conversion / gather kernels ----------------

__global__ void k_gather_half(const float* __restrict__ X, const int* __restrict__ ids,
                              _Float16* __restrict__ Xh, int n) {
    int i = blockIdx.x * blockDim.x + threadIdx.x;
    if (i >= n * HDIM) return;
    int r = i / HDIM, d = i % HDIM;
    int rr = ids ? ids[r] : r;
    Xh[i] = (_Float16)X[rr * HDIM + d];
}

// W slice [64, N] (row-major) -> Wt [N, 64] halves (so WMMA B loads are contiguous)
__global__ void k_wt_half(const float* __restrict__ W, _Float16* __restrict__ Wt, int N) {
    int i = blockIdx.x * blockDim.x + threadIdx.x;
    if (i >= N * HDIM) return;
    int nn = i / HDIM, k = i % HDIM;
    Wt[i] = (_Float16)W[k * N + nn];
}

// ---------------- WMMA GEMM: Y[M,N] = A[M,64] * W[64,N] + bias ----------------
// A: [M,64] halves row-major.  Wt: [N,64] halves (transposed W).
// One wave computes a 16x64 strip = 4 x (16x16 tiles); K = 64 = 2 x wmma(K=32).
// A fragments loaded once per strip; 8 WMMAs per wave.
// Epilogue: per-wave LDS transpose so global stores are contiguous 16B per lane.

template <bool OUT_HALF>
__global__ void k_wmma_gemm(const _Float16* __restrict__ A, const _Float16* __restrict__ Wt,
                            const float* __restrict__ bias, void* __restrict__ Yv,
                            int M, int N) {
    __shared__ float ldsBuf[8 * 256];            // one 16x16 f32 tile (or f16 tile) per wave
    const int wid  = threadIdx.x >> 5;
    const int lane = threadIdx.x & 31;
    const int wave = blockIdx.x * (blockDim.x >> 5) + wid;

    const int stripsN = N >> 6;                  // strips of 64 columns (N is 256 or 64)
    const int stripsM = (M + 15) >> 4;
    if (wave >= stripsM * stripsN) return;       // wave-uniform exit: EXEC stays all-1 for WMMA
    const int sm = wave / stripsN, sn = wave % stripsN;
    const int m0 = sm << 4, n0 = sn << 6;

    const int laneHi = lane >> 4;                // 0 or 1
    const int laneLo = lane & 15;

    // ---- A fragments (16x32 f16): elements 0..7 -> K+akoff+0..7, 8..15 -> K+akoff+16..23
    int arow = m0 + laneLo; if (arow >= M) arow = M - 1;
    const _Float16* arp = A + (size_t)arow * HDIM;
    const int akoff = laneHi << 3;
    v8h a0lo = *(const v8h*)(arp + akoff);
    v8h a0hi = *(const v8h*)(arp + akoff + 16);
    v8h a1lo = *(const v8h*)(arp + 32 + akoff);
    v8h a1hi = *(const v8h*)(arp + 32 + akoff + 16);
    v16h a0 = __builtin_shufflevector(a0lo, a0hi, 0,1,2,3,4,5,6,7,8,9,10,11,12,13,14,15);
    v16h a1 = __builtin_shufflevector(a1lo, a1hi, 0,1,2,3,4,5,6,7,8,9,10,11,12,13,14,15);

    // ---- 4 accumulators, B fragments streamed (lane = column, K+bkoff+0..7 and K+bkoff+16..23 per lane)
    const int bkoff = laneHi << 3;
    v8f c[4];
    v8f zero = {};
#pragma unroll
    for (int j = 0; j < 4; ++j) c[j] = zero;

#pragma unroll
    for (int j = 0; j < 4; ++j) {
        const _Float16* brp = Wt + (size_t)(n0 + (j << 4) + laneLo) * HDIM;
        v16h b0 = __builtin_shufflevector(*(const v8h*)(brp + bkoff), *(const v8h*)(brp + bkoff + 16), 0,1,2,3,4,5,6,7,8,9,10,11,12,13,14,15);
        v16h b1 = __builtin_shufflevector(*(const v8h*)(brp + 32 + bkoff), *(const v8h*)(brp + 32 + bkoff + 16), 0,1,2,3,4,5,6,7,8,9,10,11,12,13,14,15);
        c[j] = __builtin_amdgcn_wmma_f32_16x16x32_f16(false, a0, false, b0,
                                                      (short)0, c[j], false, false);
        c[j] = __builtin_amdgcn_wmma_f32_16x16x32_f16(false, a1, false, b1,
                                                      (short)0, c[j], false, false);
    }

    // ---- epilogue: bias add + LDS transpose + coalesced stores
    const int rr = lane >> 1;                    // readback row within tile
    const int cc = (lane & 1) << 3;              // readback col offset (0 or 8)
    const int grow = m0 + rr;

#pragma unroll
    for (int j = 0; j < 4; ++j) {
        const int colj = n0 + (j << 4) + laneLo;
        const float bv = bias[colj];
        if (OUT_HALF) {
            _Float16* lt = (_Float16*)(ldsBuf + wid * 256);
#pragma unroll
            for (int e = 0; e < 8; ++e)
                lt[(e + (laneHi << 3)) * 16 + laneLo] = (_Float16)(c[j][e] + bv);
            // same-wave DS ops are in-order; no barrier needed
            v8h o = *(const v8h*)(lt + rr * 16 + cc);
            if (grow < M)
                *(v8h*)((_Float16*)Yv + (size_t)grow * N + n0 + (j << 4) + cc) = o;
        } else {
            float* lt = ldsBuf + wid * 256;
#pragma unroll
            for (int e = 0; e < 8; ++e)
                lt[(e + (laneHi << 3)) * 16 + laneLo] = c[j][e] + bv;
            v4f o0 = *(const v4f*)(lt + rr * 16 + cc);
            v4f o1 = *(const v4f*)(lt + rr * 16 + cc + 4);
            if (grow < M) {
                float* yp = (float*)Yv + (size_t)grow * N + n0 + (j << 4) + cc;
                *(v4f*)yp = o0;
                *(v4f*)(yp + 4) = o1;
            }
        }
    }
}

// ---------------- edge (segment softmax) pipeline ----------------

__global__ void k_init(unsigned* __restrict__ menc, float* __restrict__ denom,
                       float* __restrict__ out, int ndst) {
    int i = blockIdx.x * blockDim.x + threadIdx.x;
    if (i < ndst * NHEAD) { menc[i] = NEG_INF_ENC; denom[i] = 0.0f; }
    if (i < ndst * NHEAD * HDIM) out[i] = 0.0f;
}

__global__ void k_score(const _Float16* __restrict__ Q, const _Float16* __restrict__ K,
                        const int* __restrict__ esrc, const int* __restrict__ edst,
                        float* __restrict__ score, unsigned* __restrict__ menc, int E) {
    int i = blockIdx.x * blockDim.x + threadIdx.x;
    if (i >= E * NHEAD) return;
    int e = i >> 2, h = i & 3;
    int s = esrc[e], d = edst[e];
    const _Float16* qp = Q + ((size_t)d * NHEAD + h) * HDIM;
    const _Float16* kp = K + ((size_t)s * NHEAD + h) * HDIM;
    float acc = 0.0f;
#pragma unroll
    for (int j = 0; j < HDIM; j += 8) {
        v8h qv = *(const v8h*)(qp + j);
        v8h kv = *(const v8h*)(kp + j);
#pragma unroll
        for (int t = 0; t < 8; ++t) acc += (float)qv[t] * (float)kv[t];
    }
    acc *= 0.125f;                 // 1/sqrt(64)
    score[i] = acc;
    atomicMax(menc + (size_t)d * NHEAD + h, encOrd(acc));
}

__global__ void k_expsum(const int* __restrict__ edst, float* __restrict__ score,
                         const unsigned* __restrict__ menc, float* __restrict__ denom, int E) {
    int i = blockIdx.x * blockDim.x + threadIdx.x;
    if (i >= E * NHEAD) return;
    int e = i >> 2, h = i & 3;
    int d = edst[e];
    float m = decOrd(menc[(size_t)d * NHEAD + h]);
    float ex = __expf(score[i] - m);
    score[i] = ex;
    atomicAdd(denom + (size_t)d * NHEAD + h, ex);
}

__global__ void k_scatter(const _Float16* __restrict__ V,
                          const int* __restrict__ esrc, const int* __restrict__ edst,
                          const float* __restrict__ score, const float* __restrict__ denom,
                          float* __restrict__ out, int E) {
    int i = blockIdx.x * blockDim.x + threadIdx.x;
    if (i >= E * NHEAD * 8) return;
    int g = i & 7;
    int eh = i >> 3;
    int e = eh >> 2, h = eh & 3;
    int s = esrc[e], d = edst[e];
    float w = score[eh] / denom[(size_t)d * NHEAD + h];
    const _Float16* vp = V + ((size_t)s * NHEAD + h) * HDIM + g * 8;
    float* op = out + ((size_t)d * NHEAD + h) * HDIM + g * 8;
    v8h vv = *(const v8h*)vp;
#pragma unroll
    for (int t = 0; t < 8; ++t) atomicAdd(op + t, w * (float)vv[t]);
}

__global__ void k_finalize(const float* __restrict__ out, const float* __restrict__ skip,
                           float* __restrict__ xnew, int n, int doRelu) {
    int i = blockIdx.x * blockDim.x + threadIdx.x;
    if (i >= n * HDIM) return;
    int r = i / HDIM, d = i % HDIM;
    float s = 0.0f;
#pragma unroll
    for (int h = 0; h < NHEAD; ++h) s += out[((size_t)r * NHEAD + h) * HDIM + d];
    float v = s * 0.25f + skip[i];
    if (doRelu) v = fmaxf(v, 0.0f);
    xnew[i] = v;
}

__global__ void k_pred(const float* __restrict__ xs, const float* __restrict__ xt,
                       const int* __restrict__ lbl, float* __restrict__ outp, int n) {
    int i = blockIdx.x * blockDim.x + threadIdx.x;
    if (i >= n) return;
    int a = lbl[i], b = lbl[n + i];
    const float* xa = xs + (size_t)a * HDIM;
    const float* xb = xt + (size_t)b * HDIM;
    float acc = 0.0f;
#pragma unroll
    for (int d = 0; d < HDIM; d += 4) {
        v4f va = *(const v4f*)(xa + d);
        v4f vb = *(const v4f*)(xb + d);
        acc += va.x * vb.x + va.y * vb.y + va.z * vb.z + va.w * vb.w;
    }
    outp[i] = acc;
}

// ---------------- host orchestration ----------------

extern "C" void kernel_launch(void* const* d_in, const int* in_sizes, int n_in,
                              void* d_out, int out_size, void* d_ws, size_t ws_size,
                              hipStream_t stream) {
    (void)in_sizes; (void)n_in; (void)out_size; (void)ws_size;

    const float* src_emb = (const float*)d_in[0];
    const float* tgt_emb = (const float*)d_in[1];
    const float* Wq = (const float*)d_in[2];
    const float* bq = (const float*)d_in[3];
    const float* Wk = (const float*)d_in[4];
    const float* bk = (const float*)d_in[5];
    const float* Wv = (const float*)d_in[6];
    const float* bv = (const float*)d_in[7];
    const float* Ws = (const float*)d_in[8];
    const float* bs = (const float*)d_in[9];
    const int* nid_s = (const int*)d_in[10];
    const int* nid_t = (const int*)d_in[11];
    const int* est = (const int*)d_in[12];
    const int* ets = (const int*)d_in[13];
    const int* elbl = (const int*)d_in[14];
    float* pred = (float*)d_out;

    const int NS = 50000, NT = 10000, NE = 250000, NL = 200000;

    char* ws = (char*)d_ws;
    size_t off = 0;
    auto alloc = [&](size_t bytes) -> void* {
        void* p = ws + off;
        off = (off + bytes + 255) & ~(size_t)255;
        return p;
    };

    _Float16* Xs_h = (_Float16*)alloc((size_t)NS * 64 * 2);
    _Float16* Xt_h = (_Float16*)alloc((size_t)NT * 64 * 2);
    _Float16* Wqh  = (_Float16*)alloc((size_t)4 * 256 * 64 * 2);
    _Float16* Wkh  = (_Float16*)alloc((size_t)4 * 256 * 64 * 2);
    _Float16* Wvh  = (_Float16*)alloc((size_t)4 * 256 * 64 * 2);
    _Float16* Wsh  = (_Float16*)alloc((size_t)4 * 64 * 64 * 2);
    _Float16* Qh   = (_Float16*)alloc((size_t)NS * 256 * 2);
    _Float16* Kh   = (_Float16*)alloc((size_t)NS * 256 * 2);
    _Float16* Vh   = (_Float16*)alloc((size_t)NS * 256 * 2);
    float* Skip    = (float*)alloc((size_t)NS * 64 * 4);
    float* OutAcc  = (float*)alloc((size_t)NS * 256 * 4);
    float* Score   = (float*)alloc((size_t)NE * 4 * 4);
    unsigned* Menc = (unsigned*)alloc((size_t)NS * 4 * 4);
    float* Denom   = (float*)alloc((size_t)NS * 4 * 4);
    float* xsA     = (float*)alloc((size_t)NS * 64 * 4);
    float* xtA     = (float*)alloc((size_t)NT * 64 * 4);
    float* xsB     = (float*)alloc((size_t)NS * 64 * 4);
    float* xtB     = (float*)alloc((size_t)NT * 64 * 4);

    const int B = 256;
    auto blocks = [](long n) { return (int)((n + 255) / 256); };

    // weight conversion + transpose (all layer/type slices)
    for (int sl = 0; sl < 4; ++sl) {
        k_wt_half<<<blocks(256 * 64), B, 0, stream>>>(Wq + (size_t)sl * 64 * 256, Wqh + (size_t)sl * 256 * 64, 256);
        k_wt_half<<<blocks(256 * 64), B, 0, stream>>>(Wk + (size_t)sl * 64 * 256, Wkh + (size_t)sl * 256 * 64, 256);
        k_wt_half<<<blocks(256 * 64), B, 0, stream>>>(Wv + (size_t)sl * 64 * 256, Wvh + (size_t)sl * 256 * 64, 256);
        k_wt_half<<<blocks(64 * 64),  B, 0, stream>>>(Ws + (size_t)sl * 64 * 64,  Wsh + (size_t)sl * 64 * 64,  64);
    }

    auto gemmH = [&](const _Float16* A, const _Float16* Wt, const float* bias,
                     _Float16* Y, int M, int N) {
        int strips = ((M + 15) / 16) * (N >> 6);
        k_wmma_gemm<true><<<(strips + 7) / 8, B, 0, stream>>>(A, Wt, bias, (void*)Y, M, N);
    };
    auto gemmF = [&](const _Float16* A, const _Float16* Wt, const float* bias,
                     float* Y, int M, int N) {
        int strips = ((M + 15) / 16) * (N >> 6);
        k_wmma_gemm<false><<<(strips + 7) / 8, B, 0, stream>>>(A, Wt, bias, (void*)Y, M, N);
    };

    auto runDir = [&](int l, int t, const _Float16* XsrcH, const _Float16* XdstH,
                      int Nsrc, int Ndst, const int* eidx, float* xout) {
        int sl = l * 2 + t;
        gemmH(XdstH, Wqh + (size_t)sl * 256 * 64, bq + (size_t)sl * 256, Qh, Ndst, 256);
        gemmH(XsrcH, Wkh + (size_t)sl * 256 * 64, bk + (size_t)sl * 256, Kh, Nsrc, 256);
        gemmH(XsrcH, Wvh + (size_t)sl * 256 * 64, bv + (size_t)sl * 256, Vh, Nsrc, 256);
        gemmF(XdstH, Wsh + (size_t)sl * 64 * 64,  bs + (size_t)sl * 64,  Skip, Ndst, 64);

        k_init<<<blocks((long)Ndst * 256), B, 0, stream>>>(Menc, Denom, OutAcc, Ndst);
        const int* esrc = eidx;
        const int* edst = eidx + NE;
        k_score<<<blocks((long)NE * 4), B, 0, stream>>>(Qh, Kh, esrc, edst, Score, Menc, NE);
        k_expsum<<<blocks((long)NE * 4), B, 0, stream>>>(edst, Score, Menc, Denom, NE);
        k_scatter<<<blocks((long)NE * 32), B, 0, stream>>>(Vh, esrc, edst, Score, Denom, OutAcc, NE);
        k_finalize<<<blocks((long)Ndst * 64), B, 0, stream>>>(OutAcc, Skip, xout, Ndst, l == 0 ? 1 : 0);
    };

    // ---- layer 0 ----
    k_gather_half<<<blocks((long)NS * 64), B, 0, stream>>>(src_emb, nid_s, Xs_h, NS);
    k_gather_half<<<blocks((long)NT * 64), B, 0, stream>>>(tgt_emb, nid_t, Xt_h, NT);
    runDir(0, 0, Xs_h, Xt_h, NS, NT, est, xtA);   // source -> target
    runDir(0, 1, Xt_h, Xs_h, NT, NS, ets, xsA);   // target -> source

    // ---- layer 1 ----
    k_gather_half<<<blocks((long)NS * 64), B, 0, stream>>>(xsA, nullptr, Xs_h, NS);
    k_gather_half<<<blocks((long)NT * 64), B, 0, stream>>>(xtA, nullptr, Xt_h, NT);
    runDir(1, 0, Xs_h, Xt_h, NS, NT, est, xtB);
    runDir(1, 1, Xt_h, Xs_h, NT, NS, ets, xsB);

    // ---- link prediction ----
    k_pred<<<blocks((long)NL), B, 0, stream>>>(xsB, xtB, elbl, pred, NL);
}
